// LanguageDetectorRNN_64733747085373
// MI455X (gfx1250) — compile-verified, matrix operand order checked by reading, not run on hardware
//
#include <hip/hip_runtime.h>
#include <hip/hip_bf16.h>
#include <math.h>

// ---------------------------------------------------------------------------
// MI455X (gfx1250) LSTM:
//  - bf16 WMMA (v_wmma_f32_16x16x32_bf16) for input GEMM and recurrent GEMM
//  - recurrent weights resident in VGPRs (no per-step weight traffic)
//  - TDM (tensor_load_to_lds) double-buffers xg[t] tiles into LDS
//  - cell state kept in WMMA accumulator layout (f32x8 per wave)
// ---------------------------------------------------------------------------

typedef __bf16 bf16;
typedef __attribute__((ext_vector_type(16))) __bf16 bf16x16;
typedef __attribute__((ext_vector_type(8)))  float  f32x8;
typedef __attribute__((ext_vector_type(4)))  unsigned int u32x4;
typedef __attribute__((ext_vector_type(8)))  int    i32x8;
typedef __attribute__((ext_vector_type(4)))  int    i32x4;

#define L_B 64
#define L_T 2048
#define L_H 256
#define L_G 1024              // 4*H gate rows
#define L_R (L_B * L_T)       // 131072 flattened rows
#define HPAD 264              // padded LDS row for h (bank-conflict relief)
#define SCAN_SMEM (2 * 16 * L_G * (int)sizeof(bf16) + 16 * HPAD * (int)sizeof(bf16))

#if __has_builtin(__builtin_amdgcn_tensor_load_to_lds)
#define HAVE_TDM 1
#endif

__device__ __forceinline__ f32x8 wmma_bf16(bf16x16 a, bf16x16 b, f32x8 c) {
  // 8 args: (neg_a, A, neg_b, B, c_mod, C, reuse_a, reuse_b)
  return __builtin_amdgcn_wmma_f32_16x16x32_bf16(false, a, false, b, (short)0, c,
                                                 false, false);
}

// Load a 16(row) x 32(k) bf16 tile into the WMMA A/B operand layout.
// Rows are the leading index with stride `ld` elements; k is contiguous.
// Layout per CDNA5 ISA 7.12.2 (16-bit A 16x32): lane&15 = row,
// VGPR r holds k-pairs {base, base+1} with base = (r>=4)*16 + (lane>=16)*8 + (r&3)*2.
__device__ __forceinline__ bf16x16 load_tile16(const bf16* p, int ld, int lane) {
  const int row = lane & 15;
  const int hi  = (lane >> 4) & 1;
  bf16x16 v;
#pragma unroll
  for (int r = 0; r < 8; ++r) {
    const int k = ((r >> 2) << 4) + (hi << 3) + ((r & 3) << 1);
    const bf16* q = p + (size_t)row * ld + k;
    v[2 * r]     = q[0];
    v[2 * r + 1] = q[1];
  }
  return v;
}

__device__ __forceinline__ float sigmoidf_(float x) {
  return 1.0f / (1.0f + __expf(-x));
}

#ifdef HAVE_TDM
// Build a Tensor DMA descriptor (D#) for a 2D tile:
//   tile_dim0 = 1024 bf16 (one row of gate pre-activations, contiguous)
//   tile_dim1 = 16 rows (batch rows), row stride = L_T*L_G elements
// and issue tensor_load_to_lds (clang-23 6-arg form). Tracked by TENSORcnt.
__device__ __forceinline__ void tdm_load_tile(const bf16* gsrc, unsigned lds_off) {
  const unsigned long long ga = (unsigned long long)(size_t)gsrc;
  u32x4 g0;
  g0[0] = 1u;                                            // count=1 (valid), no gather
  g0[1] = lds_off;                                       // lds_addr
  g0[2] = (unsigned)ga;                                  // global_addr[31:0]
  g0[3] = ((unsigned)(ga >> 32) & 0x1FFFFFFu) | 0x80000000u; // addr[56:32] | type=2
  i32x8 g1;
  g1[0] = (int)(1u << 16);                               // data_size=1 -> 2 bytes
  g1[1] = (int)((unsigned)(L_G & 0xFFFF) << 16);         // tensor_dim0 lo16 (=1024)
  g1[2] = (int)(16u << 16);                              // dim0 hi16=0 | tensor_dim1=16
  g1[3] = (int)((unsigned)L_G << 16);                    // dim1 hi16=0 | tile_dim0=1024
  g1[4] = 16;                                            // tile_dim1=16, tile_dim2=0
  g1[5] = (int)((unsigned)L_T * (unsigned)L_G);          // tensor_dim0_stride lo32
  g1[6] = 0;                                             // stride hi / dim1_stride lo
  g1[7] = 0;
  i32x4 g2 = {0, 0, 0, 0};                               // group 2 (unused, 2D tile)
  i32x4 g3 = {0, 0, 0, 0};                               // group 3 (unused, 2D tile)
  i32x8 g4 = {0, 0, 0, 0, 0, 0, 0, 0};                   // extra group (unused)
  __builtin_amdgcn_tensor_load_to_lds(g0, g1, g2, g3, g4, 0);
}
#endif

// ---------------------------------------------------------------------------
// fp32 -> bf16 conversion
// ---------------------------------------------------------------------------
__global__ __launch_bounds__(256) void cvt_kernel(const float* __restrict__ src,
                                                  bf16* __restrict__ dst, size_t n) {
  size_t i = (size_t)blockIdx.x * 256 + threadIdx.x;
  if (i < n) dst[i] = (bf16)src[i];
}

__global__ __launch_bounds__(256) void bias_kernel(const float* __restrict__ a,
                                                   const float* __restrict__ b,
                                                   float* __restrict__ o, int n) {
  int i = blockIdx.x * 256 + threadIdx.x;
  if (i < n) o[i] = a[i] + b[i];
}

// ---------------------------------------------------------------------------
// Phase A: xg[r, :] = x[r, :] @ w_ih^T + (b_ih + b_hh), stored bf16.
// Grid: (L_R/16, 4), 128 threads (4 waves). Each wave: one M-tile x 64 N cols.
// ---------------------------------------------------------------------------
__global__ __launch_bounds__(128) void gemm_xg_kernel(const bf16* __restrict__ xb,
                                                      const bf16* __restrict__ w,
                                                      const float* __restrict__ bias,
                                                      bf16* __restrict__ xg) {
  const int wave = threadIdx.x >> 5;
  const int lane = threadIdx.x & 31;
  const size_t m0 = (size_t)blockIdx.x * 16;
  const int ngrp = (blockIdx.y * 4 + wave) * 64;
  const int nl = lane & 15;
  const int mh = (lane >> 4) << 3;

  bf16x16 a[8];
#pragma unroll
  for (int ks = 0; ks < 8; ++ks)
    a[ks] = load_tile16(xb + m0 * L_H + ks * 32, L_H, lane);

#pragma unroll
  for (int nt = 0; nt < 4; ++nt) {
    const int n0 = ngrp + nt * 16;
    const float bv = bias[n0 + nl];
    f32x8 acc;
#pragma unroll
    for (int r = 0; r < 8; ++r) acc[r] = bv;
#pragma unroll
    for (int ks = 0; ks < 8; ++ks) {
      bf16x16 b = load_tile16(w + (size_t)n0 * L_H + ks * 32, L_H, lane);
      acc = wmma_bf16(a[ks], b, acc);
    }
#pragma unroll
    for (int r = 0; r < 8; ++r) {
      const int m = mh + r;
      xg[(m0 + m) * L_G + n0 + nl] = (bf16)acc[r];
    }
  }
}

// ---------------------------------------------------------------------------
// Phase B: recurrent scan. 4 workgroups (batch groups of 16), 16 waves each.
// Wave w owns hidden slice [16w, 16w+16): its 4 accumulators are exactly the
// i/f/g/o gate tiles for those units, so the LSTM cell update is wave-local
// and c stays in the accumulator register layout for all 2048 steps.
// w_hh (bf16, 4x8 B-operand tiles = 256 VGPRs) is preloaded into registers.
// xg[t] tiles are TDM double-buffered into LDS.
// ---------------------------------------------------------------------------
__global__ __launch_bounds__(512) void lstm_scan_kernel(const bf16* __restrict__ xg,
                                                        const bf16* __restrict__ whh,
                                                        bf16* __restrict__ y) {
  extern __shared__ char smem[];
  bf16* xgt  = (bf16*)smem;                                    // 2 x [16 x 1024]
  bf16* hbuf = (bf16*)(smem + 2 * 16 * L_G * sizeof(bf16));    // [16 x HPAD]

  const int wave = threadIdx.x >> 5;
  const int lane = threadIdx.x & 31;
  const int bg = blockIdx.x;
  const int n0 = wave * 16;       // hidden slice base
  const int nl = lane & 15;
  const int mh = (lane >> 4) << 3;

  // Preload recurrent weights into WMMA B-operand registers (resident).
  bf16x16 Breg[4][8];
#pragma unroll
  for (int g = 0; g < 4; ++g)
#pragma unroll
    for (int ks = 0; ks < 8; ++ks)
      Breg[g][ks] = load_tile16(whh + (size_t)(g * L_H + n0) * L_H + ks * 32,
                                L_H, lane);

  for (int i = threadIdx.x; i < 16 * HPAD; i += 512) hbuf[i] = (bf16)0.0f;
  f32x8 creg;
#pragma unroll
  for (int r = 0; r < 8; ++r) creg[r] = 0.0f;

  const bf16* gbase = xg + (size_t)bg * 16 * L_T * L_G;  // row (b=bg*16, t=0)

#if HAVE_TDM
  if (wave == 0) tdm_load_tile(gbase, 0u);               // prefetch t=0 -> buf0
#endif
  __syncthreads();                                       // h zeroed

  for (int t = 0; t < L_T; ++t) {
    const int cur = t & 1;
#if HAVE_TDM
    if (wave == 0) __builtin_amdgcn_s_wait_tensorcnt(0); // buf[cur] landed
#else
    {  // fallback: cooperative global->LDS copy of tile t
      const bf16* src = gbase + (size_t)t * L_G;
      unsigned* d = (unsigned*)(xgt + cur * 16 * L_G);
      for (int i = threadIdx.x; i < 16 * (L_G / 2); i += 512) {
        const int row = i / (L_G / 2), col = i % (L_G / 2);
        d[i] = ((const unsigned*)(src + (size_t)row * L_T * L_G))[col];
      }
    }
#endif
    __syncthreads();  // xg[t] visible to all waves; h from t-1 visible

    // acc = xg[t] slice (C-matrix layout: VGPR r -> m, lane -> n)
    const bf16* xrow = xgt + cur * 16 * L_G;
    f32x8 acc[4];
#pragma unroll
    for (int g = 0; g < 4; ++g)
#pragma unroll
      for (int r = 0; r < 8; ++r)
        acc[g][r] = (float)xrow[(mh + r) * L_G + g * L_H + n0 + nl];

#if HAVE_TDM
    if (wave == 0 && t + 1 < L_T)  // prefetch next tile into the other buffer
      tdm_load_tile(gbase + (size_t)(t + 1) * L_G,
                    (unsigned)((1 - cur) * 16 * L_G * sizeof(bf16)));
#endif

    // gates += h_{t-1} @ w_hh^T : 32 bf16 WMMAs on the step critical path
#pragma unroll
    for (int ks = 0; ks < 8; ++ks) {
      bf16x16 a = load_tile16(hbuf + ks * 32, HPAD, lane);
#pragma unroll
      for (int g = 0; g < 4; ++g) acc[g] = wmma_bf16(a, Breg[g][ks], acc[g]);
    }

    // LSTM cell update, all wave-local (i,f,g,o share (m,n) positions)
    f32x8 hreg;
#pragma unroll
    for (int r = 0; r < 8; ++r) {
      const float ig = sigmoidf_(acc[0][r]);
      const float fg = sigmoidf_(acc[1][r]);
      const float gg = tanhf(acc[2][r]);
      const float og = sigmoidf_(acc[3][r]);
      const float c = fg * creg[r] + ig * gg;
      creg[r] = c;
      hreg[r] = og * tanhf(c);
    }

    __syncthreads();  // everyone done reading hbuf for this step

#pragma unroll
    for (int r = 0; r < 8; ++r) {
      const int m = mh + r;
      const bf16 hv = (bf16)hreg[r];
      hbuf[m * HPAD + n0 + nl] = hv;                               // h_t -> LDS
      y[((size_t)(bg * 16 + m) * L_T + t) * L_H + n0 + nl] = hv;   // layer output
    }
    // next iteration's first barrier publishes hbuf before the A loads
  }
}

// ---------------------------------------------------------------------------
// Final linear on the last timestep: out[b] = y[b, T-1, :] . w_lin + b_lin
// ---------------------------------------------------------------------------
__global__ __launch_bounds__(32) void final_kernel(const bf16* __restrict__ y,
                                                   const float* __restrict__ wlin,
                                                   const float* __restrict__ blin,
                                                   float* __restrict__ out) {
  const int b = blockIdx.x;
  const int lane = threadIdx.x;
  const bf16* row = y + ((size_t)b * L_T + (L_T - 1)) * L_H;
  float s = 0.0f;
  for (int i = lane; i < L_H; i += 32) s += (float)row[i] * wlin[i];
#pragma unroll
  for (int off = 16; off > 0; off >>= 1) s += __shfl_down(s, off, 32);
  if (lane == 0) out[b] = s + blin[0];
}

// ---------------------------------------------------------------------------
extern "C" void kernel_launch(void* const* d_in, const int* in_sizes, int n_in,
                              void* d_out, int out_size, void* d_ws, size_t ws_size,
                              hipStream_t stream) {
  const float* input = (const float*)d_in[0];
  const float* w_ih  = (const float*)d_in[1];
  const float* w_hh  = (const float*)d_in[2];
  const float* b_ih  = (const float*)d_in[3];
  const float* b_hh  = (const float*)d_in[4];
  const float* w_lin = (const float*)d_in[5];
  const float* b_lin = (const float*)d_in[6];
  float* out = (float*)d_out;
  (void)in_sizes; (void)n_in; (void)out_size; (void)ws_size;

  // Workspace layout (~386 MB total):
  char* ws = (char*)d_ws;
  size_t off = 0;
  bf16* xb0   = (bf16*)(ws + off); off += (size_t)L_R * L_H * sizeof(bf16); // 64MB
  bf16* xb1   = (bf16*)(ws + off); off += (size_t)L_R * L_H * sizeof(bf16); // 64MB
  bf16* wihb  = (bf16*)(ws + off); off += (size_t)2 * L_G * L_H * sizeof(bf16);
  bf16* whhb  = (bf16*)(ws + off); off += (size_t)2 * L_G * L_H * sizeof(bf16);
  float* biasb = (float*)(ws + off); off += (size_t)2 * L_G * sizeof(float);
  bf16* xg    = (bf16*)(ws + off); off += (size_t)L_R * L_G * sizeof(bf16); // 256MB

  // Opt in to >64KB dynamic LDS for the scan kernel (gfx1250: up to 320KB/WG).
  (void)hipFuncSetAttribute((const void*)lstm_scan_kernel,
                            hipFuncAttributeMaxDynamicSharedMemorySize, SCAN_SMEM);

  const size_t nx = (size_t)L_R * L_H;
  cvt_kernel<<<(unsigned)((nx + 255) / 256), 256, 0, stream>>>(input, xb0, nx);
  const size_t nw = (size_t)2 * L_G * L_H;
  cvt_kernel<<<(unsigned)((nw + 255) / 256), 256, 0, stream>>>(w_ih, wihb, nw);
  cvt_kernel<<<(unsigned)((nw + 255) / 256), 256, 0, stream>>>(w_hh, whhb, nw);
  bias_kernel<<<(2 * L_G + 255) / 256, 256, 0, stream>>>(b_ih, b_hh, biasb, 2 * L_G);

  for (int l = 0; l < 2; ++l) {
    const bf16* xin = (l == 0) ? xb0 : xb1;
    bf16* xout      = (l == 0) ? xb1 : xb0;
    gemm_xg_kernel<<<dim3(L_R / 16, 4), 128, 0, stream>>>(
        xin, wihb + (size_t)l * L_G * L_H, biasb + (size_t)l * L_G, xg);
    lstm_scan_kernel<<<dim3(L_B / 16), 512, SCAN_SMEM, stream>>>(
        xg, whhb + (size_t)l * L_G * L_H, xout);
  }
  // After layer 1 the output sequence lives in xb0.
  final_kernel<<<dim3(L_B), 32, 0, stream>>>(xb0, w_lin, b_lin, out);
}
